// BiLSTM_39170101740256
// MI455X (gfx1250) — matrix operand order repaired and checked by reading, not compiled, not run on hardware
//
#include <hip/hip_runtime.h>
#include <hip/hip_fp16.h>

// ---- problem dims ----
#define T_DIM 2048
#define I_DIM 1024
#define H_DIM 1024
#define G_DIM 4096   // 4*H
#define H2    2048   // 2*H
#define C_DIM 1000
#define CPAD  1024   // padded FC output cols (zero-filled weights)

typedef _Float16 v16h __attribute__((ext_vector_type(16)));
typedef _Float16 v8h  __attribute__((ext_vector_type(8)));
typedef float    v8f  __attribute__((ext_vector_type(8)));

// -------------------- small utility kernels --------------------
__global__ void k_f2h(const float* __restrict__ s, _Float16* __restrict__ d, int n) {
    for (int i = blockIdx.x * blockDim.x + threadIdx.x; i < n; i += gridDim.x * blockDim.x)
        d[i] = (_Float16)s[i];
}

__global__ void k_zero_h(_Float16* __restrict__ d, int n) {
    for (int i = blockIdx.x * blockDim.x + threadIdx.x; i < n; i += gridDim.x * blockDim.x)
        d[i] = (_Float16)0.0f;
}

__global__ void k_bias_sum(const float* __restrict__ a, const float* __restrict__ b,
                           float* __restrict__ o, int n) {
    int i = blockIdx.x * blockDim.x + threadIdx.x;
    if (i < n) o[i] = a[i] + b[i];
}

__global__ void k_init_sync(int* __restrict__ s) {
    if (threadIdx.x < 8) s[threadIdx.x] = 0;
}

// -------------------- WMMA GEMM: C[m,n] = sum_k A[m,k]*B[n,k] + bias[n] ----
// A: [M,K] f16 row-major (K contiguous), B: [N,K] f16 row-major, C: f32.
// Block = 256 threads = 8 waves (2x4): block tile 32(M) x 256(N), wave tile 16x64.
// Ping-pong software pipeline: two fragment register sets X/Y; each phase
// loads fresh fragments into one set while issuing WMMAs from the other, so
// there are no rotation copies and load latency hides under the matrix ops.
// K must be a multiple of 64 (both call sites: 1024 and 2048).

__device__ __forceinline__ v16h frag_from(const _Float16* __restrict__ p, int hi_off) {
    v8h lo = *(const v8h*)(p);
    v8h hi = *(const v8h*)(p + hi_off);
    return __builtin_shufflevector(lo, hi, 0,1,2,3,4,5,6,7,8,9,10,11,12,13,14,15);
}

__global__ void k_gemm_f16(const _Float16* __restrict__ A, const _Float16* __restrict__ B,
                           const float* __restrict__ bias, float* __restrict__ C,
                           int M, int N, int K, int ldc, int nstore) {
    (void)M; (void)N;
    const int tid  = threadIdx.x;
    const int wave = tid >> 5;
    const int lane = tid & 31;
    const int wm   = wave >> 2;           // 0..1
    const int wn   = wave & 3;            // 0..3
    const int row0 = blockIdx.x * 32 + wm * 16;
    const int col0 = blockIdx.y * 256 + wn * 64;
    const int l15  = lane & 15;
    const int kh   = lane >> 4;

    // per-lane fragment pointers (A and B: 16-elem halves)
    const _Float16* pa  = A + (size_t)(row0 + l15) * K + kh * 16;
    const _Float16* pb0 = B + (size_t)(col0 +  0 + l15) * K + kh * 16;
    const _Float16* pb1 = B + (size_t)(col0 + 16 + l15) * K + kh * 16;
    const _Float16* pb2 = B + (size_t)(col0 + 32 + l15) * K + kh * 16;
    const _Float16* pb3 = B + (size_t)(col0 + 48 + l15) * K + kh * 16;

    v8f acc0 = {}, acc1 = {}, acc2 = {}, acc3 = {};

    // prologue: set X <- slice 0
    v16h xa  = frag_from(pa, 8);
    v16h xb0 = frag_from(pb0, 8);
    v16h xb1 = frag_from(pb1, 8);
    v16h xb2 = frag_from(pb2, 8);
    v16h xb3 = frag_from(pb3, 8);
    pa += 32; pb0 += 32; pb1 += 32; pb2 += 32; pb3 += 32;

    v16h ya, yb0, yb1, yb2, yb3;

    const int steady = (K / 32 - 2) / 2;
#pragma unroll 1
    for (int it = 0; it < steady; ++it) {
        // phase A: load Y, compute X
        ya  = frag_from(pa, 8);
        yb0 = frag_from(pb0, 8);
        yb1 = frag_from(pb1, 8);
        yb2 = frag_from(pb2, 8);
        yb3 = frag_from(pb3, 8);
        pa += 32; pb0 += 32; pb1 += 32; pb2 += 32; pb3 += 32;
        __builtin_prefetch((const char*)(pa + 32), 0, 1);   // global_prefetch_b8
        acc0 = __builtin_amdgcn_wmma_f32_16x16x32_f16(false, xa, false, xb0, (short)0, acc0, false, false);
        acc1 = __builtin_amdgcn_wmma_f32_16x16x32_f16(false, xa, false, xb1, (short)0, acc1, false, false);
        acc2 = __builtin_amdgcn_wmma_f32_16x16x32_f16(false, xa, false, xb2, (short)0, acc2, false, false);
        acc3 = __builtin_amdgcn_wmma_f32_16x16x32_f16(false, xa, false, xb3, (short)0, acc3, false, false);
        // phase B: load X, compute Y
        xa  = frag_from(pa, 8);
        xb0 = frag_from(pb0, 8);
        xb1 = frag_from(pb1, 8);
        xb2 = frag_from(pb2, 8);
        xb3 = frag_from(pb3, 8);
        pa += 32; pb0 += 32; pb1 += 32; pb2 += 32; pb3 += 32;
        acc0 = __builtin_amdgcn_wmma_f32_16x16x32_f16(false, ya, false, yb0, (short)0, acc0, false, false);
        acc1 = __builtin_amdgcn_wmma_f32_16x16x32_f16(false, ya, false, yb1, (short)0, acc1, false, false);
        acc2 = __builtin_amdgcn_wmma_f32_16x16x32_f16(false, ya, false, yb2, (short)0, acc2, false, false);
        acc3 = __builtin_amdgcn_wmma_f32_16x16x32_f16(false, ya, false, yb3, (short)0, acc3, false, false);
    }

    // tail: one more slice in Y, then drain both sets
    ya  = frag_from(pa, 8);
    yb0 = frag_from(pb0, 8);
    yb1 = frag_from(pb1, 8);
    yb2 = frag_from(pb2, 8);
    yb3 = frag_from(pb3, 8);
    acc0 = __builtin_amdgcn_wmma_f32_16x16x32_f16(false, xa, false, xb0, (short)0, acc0, false, false);
    acc1 = __builtin_amdgcn_wmma_f32_16x16x32_f16(false, xa, false, xb1, (short)0, acc1, false, false);
    acc2 = __builtin_amdgcn_wmma_f32_16x16x32_f16(false, xa, false, xb2, (short)0, acc2, false, false);
    acc3 = __builtin_amdgcn_wmma_f32_16x16x32_f16(false, xa, false, xb3, (short)0, acc3, false, false);
    acc0 = __builtin_amdgcn_wmma_f32_16x16x32_f16(false, ya, false, yb0, (short)0, acc0, false, false);
    acc1 = __builtin_amdgcn_wmma_f32_16x16x32_f16(false, ya, false, yb1, (short)0, acc1, false, false);
    acc2 = __builtin_amdgcn_wmma_f32_16x16x32_f16(false, ya, false, yb2, (short)0, acc2, false, false);
    acc3 = __builtin_amdgcn_wmma_f32_16x16x32_f16(false, ya, false, yb3, (short)0, acc3, false, false);

    // C/D layout: VGPR v, lanes 0-15 -> M = v; lanes 16-31 -> M = v+8; N = lane%16.
    const int rbase = row0 + (kh << 3);
    v8f accs[4] = {acc0, acc1, acc2, acc3};
#pragma unroll
    for (int t = 0; t < 4; ++t) {
        int col = col0 + t * 16 + l15;
        if (col < nstore) {
            float bv = bias[col];
#pragma unroll
            for (int v = 0; v < 8; ++v)
                C[(size_t)(rbase + v) * ldc + col] = accs[t][v] + bv;
        }
    }
}

// -------------------- persistent bidirectional LSTM recurrence --------------------
// Grid: 16 blocks x 1024 threads. Blocks 0..7 = forward dir, 8..15 = backward dir.
// Each block owns 128 hidden units j in [wg*128, wg*128+128): i.e. W_hh rows
// {g*H + j} for g in 0..3 (512 rows). Two threads per row (each 512 K elems).
// h is double-buffered in global memory; one device-scope barrier per step.
// h_prev is staged into LDS with CDNA5 async global->LDS copies (ASYNCcnt path).

#define NWG_DIR 8
#define LSTM_THREADS 1024

__device__ __forceinline__ void grid_barrier(int* cnt, int* gen, int nwg) {
    __syncthreads();
    if (threadIdx.x == 0) {
        __threadfence();
        int g = __hip_atomic_load(gen, __ATOMIC_RELAXED, __HIP_MEMORY_SCOPE_AGENT);
        int a = __hip_atomic_fetch_add(cnt, 1, __ATOMIC_ACQ_REL, __HIP_MEMORY_SCOPE_AGENT);
        if (a == nwg - 1) {
            __hip_atomic_store(cnt, 0, __ATOMIC_RELAXED, __HIP_MEMORY_SCOPE_AGENT);
            __hip_atomic_fetch_add(gen, 1, __ATOMIC_RELEASE, __HIP_MEMORY_SCOPE_AGENT);
        } else {
            while (__hip_atomic_load(gen, __ATOMIC_ACQUIRE, __HIP_MEMORY_SCOPE_AGENT) == g)
                __builtin_amdgcn_s_sleep(1);
        }
    }
    __syncthreads();
}

__device__ __forceinline__ float sigmoidf_(float x) {
    return 1.0f / (1.0f + __expf(-x));
}

__global__ void __launch_bounds__(LSTM_THREADS)
k_lstm(const float* __restrict__ xpf, const float* __restrict__ xpb,
       const float* __restrict__ whf, const float* __restrict__ whb,
       _Float16* __restrict__ hist, float* __restrict__ hbuf_all,
       int* __restrict__ sync) {
    const int dir = (blockIdx.x >= NWG_DIR) ? 1 : 0;
    const int wg  = blockIdx.x & (NWG_DIR - 1);
    const float* __restrict__ xproj = dir ? xpb : xpf;
    const float* __restrict__ Whh   = dir ? whb : whf;
    float* hb = hbuf_all + dir * (2 * H_DIM);   // [2][H] double buffer for this dir
    int* cnt = sync + dir * 2;
    int* gen = cnt + 1;

    const int tid = threadIdx.x;
    const int j0  = wg * 128;
    const int r   = tid >> 1;          // 0..511 : local row
    const int gi  = r >> 7;            // gate 0..3 (i,f,g,o)
    const int jl  = r & 127;           // unit within block slice
    const int grow = gi * H_DIM + j0 + jl;
    const int kh   = tid & 1;
    const int kbase = kh * 512;
    const float4* __restrict__ wrow = (const float4*)(Whh + (size_t)grow * H_DIM + kbase);

    __shared__ float hs[H_DIM];
    __shared__ float gsm[512];

    // per-lane LDS byte offset for the async copy of h_prev (one f32 per thread)
    const int hidx = tid & (H_DIM - 1);
    const unsigned lds_off = (unsigned)(size_t)(&hs[hidx]);

    float c = 0.0f;   // cell state, valid for tid < 128 (owner of unit j0+tid)

    // zero both h buffers for our slice, then sync the whole direction
    if (tid < 128) {
        hb[j0 + tid] = 0.0f;
        hb[H_DIM + j0 + tid] = 0.0f;
    }
    grid_barrier(cnt, gen, NWG_DIR);

    for (int s = 0; s < T_DIM; ++s) {
        const int t = dir ? (T_DIM - 1 - s) : s;
        const float* __restrict__ hprev = hb + (s & 1) * H_DIM;
        float* __restrict__ hnext = hb + ((s + 1) & 1) * H_DIM;

        // stage h_prev into LDS via CDNA5 async global->LDS copy
        __syncthreads();                     // protect hs/gsm reuse from prev iter
        {
            unsigned long long gaddr = (unsigned long long)(const void*)(hprev + hidx);
            asm volatile("global_load_async_to_lds_b32 %0, %1, off"
                         :: "v"(lds_off), "v"(gaddr) : "memory");
            asm volatile("s_wait_asynccnt 0x0" ::: "memory");
        }
        __syncthreads();

        // 512-long partial dot (fp32)
        const float4* __restrict__ hv = (const float4*)(hs + kbase);
        float acc = 0.0f;
#pragma unroll 8
        for (int i = 0; i < 128; ++i) {
            float4 w = wrow[i];
            float4 h4 = hv[i];
            acc = fmaf(w.x, h4.x, acc);
            acc = fmaf(w.y, h4.y, acc);
            acc = fmaf(w.z, h4.z, acc);
            acc = fmaf(w.w, h4.w, acc);
        }
        acc += __shfl_xor(acc, 1);           // combine the two K halves
        if (kh == 0) gsm[r] = acc;
        __syncthreads();

        if (tid < 128) {
            const int j = j0 + tid;
            const float* xp = xproj + (size_t)t * G_DIM;
            float gi_ = sigmoidf_(gsm[  0 + tid] + xp[          j]);
            float gf  = sigmoidf_(gsm[128 + tid] + xp[H_DIM   + j]);
            float gg  = tanhf    (gsm[256 + tid] + xp[2*H_DIM + j]);
            float go  = sigmoidf_(gsm[384 + tid] + xp[3*H_DIM + j]);
            c = gf * c + gi_ * gg;
            float h = go * tanhf(c);
            hnext[j] = h;
            hist[(size_t)t * H2 + dir * H_DIM + j] = (_Float16)h;
        }
        grid_barrier(cnt, gen, NWG_DIR);     // makes hnext visible to all WGs of dir
    }
}

// -------------------- launch --------------------
extern "C" void kernel_launch(void* const* d_in, const int* in_sizes, int n_in,
                              void* d_out, int out_size, void* d_ws, size_t ws_size,
                              hipStream_t stream) {
    (void)in_sizes; (void)n_in; (void)out_size; (void)ws_size;

    const float* x      = (const float*)d_in[0];
    const float* W_ih_f = (const float*)d_in[1];
    const float* W_hh_f = (const float*)d_in[2];
    const float* b_ih_f = (const float*)d_in[3];
    const float* b_hh_f = (const float*)d_in[4];
    const float* W_ih_b = (const float*)d_in[5];
    const float* W_hh_b = (const float*)d_in[6];
    const float* b_ih_b = (const float*)d_in[7];
    const float* b_hh_b = (const float*)d_in[8];
    const float* fc_W   = (const float*)d_in[9];
    const float* fc_b   = (const float*)d_in[10];
    float* out = (float*)d_out;

    // workspace carve-up (256B aligned), total ~101 MB
    char* ws = (char*)d_ws;
    size_t off = 0;
    auto take = [&](size_t bytes) -> char* {
        char* p = ws + off;
        off += (bytes + 255) & ~(size_t)255;
        return p;
    };
    _Float16* xh      = (_Float16*)take((size_t)T_DIM * I_DIM * 2);
    _Float16* wihf_h  = (_Float16*)take((size_t)G_DIM * I_DIM * 2);
    _Float16* wihb_h  = (_Float16*)take((size_t)G_DIM * I_DIM * 2);
    _Float16* fcw_h   = (_Float16*)take((size_t)CPAD * H2 * 2);
    float*    bias_f  = (float*)take((size_t)G_DIM * 4);
    float*    bias_b  = (float*)take((size_t)G_DIM * 4);
    float*    xproj_f = (float*)take((size_t)T_DIM * G_DIM * 4);
    float*    xproj_b = (float*)take((size_t)T_DIM * G_DIM * 4);
    _Float16* h_hist  = (_Float16*)take((size_t)T_DIM * H2 * 2);
    float*    hbuf    = (float*)take((size_t)2 * 2 * H_DIM * 4);
    int*      syncp   = (int*)take(256);

    auto gsz = [](size_t n) { size_t g = (n + 255) / 256; return (int)(g > 4096 ? 4096 : g); };

    // conversions + bias folding
    k_zero_h<<<gsz((size_t)CPAD * H2), 256, 0, stream>>>(fcw_h, CPAD * H2);
    k_f2h<<<gsz((size_t)T_DIM * I_DIM), 256, 0, stream>>>(x, xh, T_DIM * I_DIM);
    k_f2h<<<gsz((size_t)G_DIM * I_DIM), 256, 0, stream>>>(W_ih_f, wihf_h, G_DIM * I_DIM);
    k_f2h<<<gsz((size_t)G_DIM * I_DIM), 256, 0, stream>>>(W_ih_b, wihb_h, G_DIM * I_DIM);
    k_f2h<<<gsz((size_t)C_DIM * H2), 256, 0, stream>>>(fc_W, fcw_h, C_DIM * H2);
    k_bias_sum<<<(G_DIM + 255) / 256, 256, 0, stream>>>(b_ih_f, b_hh_f, bias_f, G_DIM);
    k_bias_sum<<<(G_DIM + 255) / 256, 256, 0, stream>>>(b_ih_b, b_hh_b, bias_b, G_DIM);
    k_init_sync<<<1, 32, 0, stream>>>(syncp);

    // x-projections (bias folded): xproj[t, n] = x[t]·W_ih[n] + b_ih[n] + b_hh[n]
    dim3 gp(T_DIM / 32, G_DIM / 256);
    k_gemm_f16<<<gp, 256, 0, stream>>>(xh, wihf_h, bias_f, xproj_f,
                                       T_DIM, G_DIM, I_DIM, G_DIM, G_DIM);
    k_gemm_f16<<<gp, 256, 0, stream>>>(xh, wihb_h, bias_b, xproj_b,
                                       T_DIM, G_DIM, I_DIM, G_DIM, G_DIM);

    // persistent bidirectional recurrence
    k_lstm<<<2 * NWG_DIR, LSTM_THREADS, 0, stream>>>(xproj_f, xproj_b, W_hh_f, W_hh_b,
                                                     h_hist, hbuf, syncp);

    // final FC: out[t, c] = h_hist[t]·fc_W[c] + fc_b[c]
    dim3 gf(T_DIM / 32, CPAD / 256);
    k_gemm_f16<<<gf, 256, 0, stream>>>(h_hist, fcw_h, fc_b, out,
                                       T_DIM, CPAD, H2, C_DIM, C_DIM);
}
